// GConvLSTM_Model_8581344657588
// MI455X (gfx1250) — matrix operand order repaired and checked by reading, not compiled, not run on hardware
//
#include <hip/hip_runtime.h>
#include <hip/hip_bf16.h>

typedef __attribute__((ext_vector_type(16))) _Float16 v16h;
typedef __attribute__((ext_vector_type(8)))  _Float16 v8h;
typedef __attribute__((ext_vector_type(8)))  float    v8f;

static constexpr int N_NODES = 50000;
static constexpr int N_EDGES = 800000;
static constexpr int F_IN    = 100;
static constexpr int H1      = 50;
static constexpr int H2      = 20;

// ---------------------------------------------------------------- utilities
__global__ void k_zero(float* __restrict__ p, long long n) {
  long long i = (long long)blockIdx.x * blockDim.x + threadIdx.x;
  if (i < n) p[i] = 0.0f;
}

// deg[src] += w  (segment_sum over source nodes)
__global__ void k_deg(const int* __restrict__ ei, const float* __restrict__ ew,
                      float* __restrict__ deg, int E) {
  int e = blockIdx.x * blockDim.x + threadIdx.x;
  if (e >= E) return;
  atomicAdd(&deg[ei[e]], ew[e]);
}

// dis = deg>0 ? rsqrt(deg) : 0   (in place)
__global__ void k_dis(float* __restrict__ d, int n) {
  int i = blockIdx.x * blockDim.x + threadIdx.x;
  if (i >= n) return;
  float v = d[i];
  d[i] = v > 0.0f ? rsqrtf(v) : 0.0f;
}

// w_hat[e] = -dis[src]*ew*dis[dst]
__global__ void k_what(const int* __restrict__ ei, const float* __restrict__ ew,
                       const float* __restrict__ dis, float* __restrict__ what, int E) {
  int e = blockIdx.x * blockDim.x + threadIdx.x;
  if (e >= E) return;
  int s = ei[e], d = ei[E + e];
  what[e] = -dis[s] * ew[e] * dis[d];
}

// out[dst,f] += w_hat[e] * v[src,f]   (L2-resident float atomics)
__global__ void k_spmm(const float* __restrict__ v, const int* __restrict__ ei,
                       const float* __restrict__ what, float* __restrict__ out,
                       int E, int F) {
  long long i = (long long)blockIdx.x * blockDim.x + threadIdx.x;
  long long total = (long long)E * F;
  if (i >= total) return;
  int e = (int)(i / F), f = (int)(i % F);
  int s = ei[e], d = ei[E + e];
  atomicAdd(&out[(size_t)d * F + f], what[e] * v[(size_t)s * F + f]);
}

// A[m, 0:F] = X ; A[m, Kp1:Kp1+F] = LX ; else 0   (f32 -> f16, zero padded)
__global__ void k_packA(const float* __restrict__ X, const float* __restrict__ LX,
                        _Float16* __restrict__ A, int M, int F, int Kp1, int Kpad) {
  long long i = (long long)blockIdx.x * blockDim.x + threadIdx.x;
  long long total = (long long)M * Kpad;
  if (i >= total) return;
  int m = (int)(i / Kpad), k = (int)(i % Kpad);
  float v = 0.0f;
  if (k < F)                        v = X[(size_t)m * F + k];
  else if (k >= Kp1 && k < Kp1 + F) v = LX[(size_t)m * F + (k - Kp1)];
  A[i] = (_Float16)v;
}

// Pack stacked gate weights [Kpad x 4*out] directly into per-lane WMMA B
// fragment layout: Bp[((nt*KC+kc)*32 + lane)*16 + j] = B[kc*32 + (j>>3)*16 + (lane>>4)*8 + (j&7)][nt*16 + (lane&15)]
// Rows 0:F = w[0]; rows Kp1:Kp1+F = w[1]; everything else zero.
__global__ void k_packB(const float* __restrict__ wi, const float* __restrict__ wf,
                        const float* __restrict__ wc, const float* __restrict__ wo,
                        int F, int out, int Kp1, int NT, int KC,
                        _Float16* __restrict__ Bp) {
  int i = blockIdx.x * blockDim.x + threadIdx.x;
  int total = NT * KC * 512;
  if (i >= total) return;
  int j    = i & 15;
  int lane = (i >> 4) & 31;
  int tile = i >> 9;
  int kc = tile % KC;
  int nt = tile / KC;
  int K = kc * 32 + (j >> 3) * 16 + (lane >> 4) * 8 + (j & 7);
  int n = nt * 16 + (lane & 15);
  float v = 0.0f;
  if (n < 4 * out) {
    int g  = n / out;
    int jj = n % out;
    const float* w = (g == 0) ? wi : (g == 1) ? wf : (g == 2) ? wc : wo;
    if (K < F)                        v = w[(size_t)K * out + jj];              // w[0]
    else if (K >= Kp1 && K < Kp1 + F) v = w[(size_t)(F + (K - Kp1)) * out + jj]; // w[1]
  }
  Bp[i] = (_Float16)v;
}

// One wave -> one 16x16 C tile. A: [M x Kpad] f16 row-major (zero padded);
// Bp: pre-swizzled fragments. C: [M x Npad] f32.
__global__ void k_gemm(const _Float16* __restrict__ A, const _Float16* __restrict__ Bp,
                       float* __restrict__ C, int MT, int NT, int KC, int Kpad, int Npad) {
  int wave = blockIdx.x * (blockDim.x >> 5) + (threadIdx.x >> 5);
  int lane = threadIdx.x & 31;
  if (wave >= MT * NT) return;               // wave-uniform: EXEC stays all-ones
  int nt = wave % NT, mt = wave / NT;
  int m0 = mt << 4, n0 = nt << 4;
  int m = lane & 15, half = lane >> 4;

  v8f acc = {};
  const _Float16* arow  = A  + (size_t)(m0 + m) * Kpad + half * 8;
  const _Float16* bfrag = Bp + (((size_t)nt * KC) * 32 + lane) * 16;

  for (int kc = 0; kc < KC; ++kc) {
    v8h alo = *(const v8h*)(arow);        // K = kbase + half*8 + 0..7
    v8h ahi = *(const v8h*)(arow + 16);   // K = kbase + half*8 + 16..23
    v8h blo = *(const v8h*)(bfrag);
    v8h bhi = *(const v8h*)(bfrag + 8);
    v16h a, b;
#pragma unroll
    for (int t = 0; t < 8; ++t) {
      a[t] = alo[t]; a[t + 8] = ahi[t];
      b[t] = blo[t]; b[t + 8] = bhi[t];
    }
    acc = __builtin_amdgcn_wmma_f32_16x16x32_f16(
        /*neg_a=*/false, a, /*neg_b=*/false, b,
        /*c_mod=*/(short)0, acc, /*reuse_a=*/false, /*reuse_b=*/false);
    arow  += 32;
    bfrag += 512;
  }
  // C/D layout: VGPR r -> M = r + half*8, N = lane&15
  float* crow = C + (size_t)(m0 + half * 8) * Npad + n0 + m;
#pragma unroll
  for (int r = 0; r < 8; ++r) crow[(size_t)r * Npad] = acc[r];
}

// Gate nonlinearity (single step, H=C=0 simplification; F-gate unused).
__global__ void k_gates(const float* __restrict__ G, int Npad, int out,
                        const float* __restrict__ bxi, const float* __restrict__ bhi, const float* __restrict__ bi,
                        const float* __restrict__ bxc, const float* __restrict__ bhc, const float* __restrict__ bc,
                        const float* __restrict__ bxo, const float* __restrict__ bho, const float* __restrict__ bo,
                        const float* __restrict__ wco, float* __restrict__ H, int Nn) {
  long long i = (long long)blockIdx.x * blockDim.x + threadIdx.x;
  long long total = (long long)Nn * out;
  if (i >= total) return;
  int n = (int)(i / out), j = (int)(i % out);
  const float* g = G + (size_t)n * Npad;
  float Pi = g[j]           + bxi[j] + bhi[j] + bi[j];
  float Pc = g[2 * out + j] + bxc[j] + bhc[j] + bc[j];
  float Po = g[3 * out + j] + bxo[j] + bho[j] + bo[j];
  float I  = 1.0f / (1.0f + __expf(-Pi));
  float T  = tanhf(Pc);
  float Cc = I * T;
  float O  = 1.0f / (1.0f + __expf(-(Po + wco[j] * Cc)));
  float h  = O * tanhf(Cc);
  H[i] = h > 0.0f ? h : 0.0f;
}

__global__ void k_linear(const float* __restrict__ H, const float* __restrict__ w,
                         const float* __restrict__ b, float* __restrict__ out,
                         int Nn, int F) {
  int n = blockIdx.x * blockDim.x + threadIdx.x;
  if (n >= Nn) return;
  float acc = b[0];
#pragma unroll
  for (int j = 0; j < H2; ++j) acc += H[(size_t)n * F + j] * w[j];
  out[n] = acc;
}

// ---------------------------------------------------------------- launch
extern "C" void kernel_launch(void* const* d_in, const int* in_sizes, int n_in,
                              void* d_out, int out_size, void* d_ws, size_t ws_size,
                              hipStream_t stream) {
  (void)in_sizes; (void)n_in; (void)out_size; (void)ws_size;

  int p = 0;
  auto f = [&](int idx) { return (const float*)d_in[idx]; };

  const float* x  = f(p++);                 // [50000,100]
  const int*   ei = (const int*)d_in[p++];  // [2,800000]
  const float* ew = f(p++);                 // [800000]

  struct Layer {
    const float *xw[4], *xb[4], *hw[4], *hb[4], *bg[4];
    const float *wci, *wcf, *wco;
  };
  auto readLayer = [&](Layer& L) {
    for (int g = 0; g < 4; ++g) {           // insertion order: i, f, c, o
      L.xw[g] = f(p++); L.xb[g] = f(p++);   // x_g: {w, b}
      L.hw[g] = f(p++); L.hb[g] = f(p++);   // h_g: {w, b}
      L.bg[g] = f(p++);                     // b_g
    }
    L.wci = f(p++); L.wcf = f(p++); L.wco = f(p++);
  };
  Layer L1, L2;
  readLayer(L1);
  readLayer(L2);
  const float* lin_w = f(p++);
  const float* lin_b = f(p++);

  // -------- workspace layout (all L2-resident: ~105 MB total)
  char* ws = (char*)d_ws;
  size_t off = 0;
  auto alloc = [&](size_t bytes) {
    size_t o = off;
    off += (bytes + 255) & ~(size_t)255;
    return o;
  };
  float*    deg  = (float*)   (ws + alloc((size_t)N_NODES * 4));
  float*    what = (float*)   (ws + alloc((size_t)N_EDGES * 4));
  float*    lap  = (float*)   (ws + alloc((size_t)N_NODES * F_IN * 4));   // reused for L*h1
  _Float16* A    = (_Float16*)(ws + alloc((size_t)N_NODES * 256 * 2));    // reused layer2
  _Float16* Bp   = (_Float16*)(ws + alloc((size_t)13 * 8 * 512 * 2));     // reused layer2
  float*    G    = (float*)   (ws + alloc((size_t)N_NODES * 208 * 4));    // reused layer2
  float*    h1   = (float*)   (ws + alloc((size_t)N_NODES * H1 * 4));
  float*    h2   = (float*)   (ws + alloc((size_t)N_NODES * H2 * 4));

  const int B = 256;
  auto cd = [](long long a, int b) { return (int)((a + b - 1) / b); };

  // -------- graph normalization
  k_zero<<<cd(N_NODES, B), B, 0, stream>>>(deg, N_NODES);
  k_deg <<<cd(N_EDGES, B), B, 0, stream>>>(ei, ew, deg, N_EDGES);
  k_dis <<<cd(N_NODES, B), B, 0, stream>>>(deg, N_NODES);
  k_what<<<cd(N_EDGES, B), B, 0, stream>>>(ei, ew, deg, what, N_EDGES);

  // -------- layer 1: L*x, pack, WMMA GEMM, gates
  k_zero <<<cd((long long)N_NODES * F_IN, B), B, 0, stream>>>(lap, (long long)N_NODES * F_IN);
  k_spmm <<<cd((long long)N_EDGES * F_IN, B), B, 0, stream>>>(x, ei, what, lap, N_EDGES, F_IN);
  k_packA<<<cd((long long)N_NODES * 256, B), B, 0, stream>>>(x, lap, A, N_NODES, F_IN, 128, 256);
  k_packB<<<cd(13 * 8 * 512, B), B, 0, stream>>>(L1.xw[0], L1.xw[1], L1.xw[2], L1.xw[3],
                                                 F_IN, H1, 128, 13, 8, Bp);
  {
    int MT = N_NODES / 16, NT = 13;
    k_gemm<<<cd((long long)MT * NT, 8), B, 0, stream>>>(A, Bp, G, MT, NT, 8, 256, 208);
  }
  k_gates<<<cd((long long)N_NODES * H1, B), B, 0, stream>>>(
      G, 208, H1,
      L1.xb[0], L1.hb[0], L1.bg[0],
      L1.xb[2], L1.hb[2], L1.bg[2],
      L1.xb[3], L1.hb[3], L1.bg[3],
      L1.wco, h1, N_NODES);

  // -------- layer 2: L*h1, pack, WMMA GEMM, gates
  k_zero <<<cd((long long)N_NODES * H1, B), B, 0, stream>>>(lap, (long long)N_NODES * H1);
  k_spmm <<<cd((long long)N_EDGES * H1, B), B, 0, stream>>>(h1, ei, what, lap, N_EDGES, H1);
  k_packA<<<cd((long long)N_NODES * 128, B), B, 0, stream>>>(h1, lap, A, N_NODES, H1, 64, 128);
  k_packB<<<cd(5 * 4 * 512, B), B, 0, stream>>>(L2.xw[0], L2.xw[1], L2.xw[2], L2.xw[3],
                                                H1, H2, 64, 5, 4, Bp);
  {
    int MT = N_NODES / 16, NT = 5;
    k_gemm<<<cd((long long)MT * NT, 8), B, 0, stream>>>(A, Bp, G, MT, NT, 4, 128, 80);
  }
  k_gates<<<cd((long long)N_NODES * H2, B), B, 0, stream>>>(
      G, 80, H2,
      L2.xb[0], L2.hb[0], L2.bg[0],
      L2.xb[2], L2.hb[2], L2.bg[2],
      L2.xb[3], L2.hb[3], L2.bg[3],
      L2.wco, h2, N_NODES);

  // -------- readout
  k_linear<<<cd(N_NODES, B), B, 0, stream>>>(h2, lin_w, lin_b, (float*)d_out, N_NODES, H2);
}
